// PureLRUR_34857954574991
// MI455X (gfx1250) — matrix operand order repaired and checked by reading, not compiled, not run on hardware
//
#include <hip/hip_runtime.h>
#include <hip/hip_bf16.h>
#include <math.h>

// Problem sizes (match reference)
#define BATCH   16
#define SEQ     4096
#define H_IN    128
#define N_STATE 256
#define H_OUT   128
#define BL      (BATCH * SEQ)       // 65536 rows
#define CHUNK   64
#define NCHUNK  (SEQ / CHUNK)       // 64

// Activation buffer row layout (bf16): [pre_re(256) | pre_im(256) | u(128)] = 640
#define AK      640
#define MT      2                   // M-tiles (of 16 rows) per wave

typedef __attribute__((ext_vector_type(16))) __bf16 v16bf;
typedef __attribute__((ext_vector_type(8)))  __bf16 v8bf;
typedef __attribute__((ext_vector_type(8)))  float  v8f;
typedef __attribute__((ext_vector_type(4)))  unsigned int u32x4;
typedef __attribute__((ext_vector_type(8)))  int i32x8;
typedef __attribute__((ext_vector_type(4)))  int i32x4;

// ---------------------------------------------------------------------------
// WMMA fragment loaders (CDNA5 16-bit layouts, ISA 7.12.2)
// A 16x32 bf16: lanes 0-15 row M=lane, elems = K{0..7,16..23}; lanes 16-31
// row M=lane-16, elems = K{8..15,24..31}.
// B 32x16 bf16 (stored [N][K] row-major): lanes 0-15 col N=lane hold
// K{0..7,16..23}; lanes 16-31 col N=lane-16 hold K{8..15,24..31}.
// ---------------------------------------------------------------------------
__device__ __forceinline__ v16bf load_frag_a(const __bf16* base, int lda,
                                             int m0, int k0, int lane) {
    int half = lane >> 4;
    int mr = m0 + (lane & 15);
    const __bf16* p = base + (size_t)mr * lda + k0 + half * 8;
    union { v16bf v; v8bf h[2]; } u;
    u.h[0] = *(const v8bf*)(p);
    u.h[1] = *(const v8bf*)(p + 16);
    return u.v;
}

__device__ __forceinline__ v16bf lds_frag_b(const __bf16* wl, int K,
                                            int n0, int k0, int lane) {
    int half = lane >> 4;
    int nc = n0 + (lane & 15);
    const __bf16* p = wl + nc * K + k0 + half * 8;
    union { v16bf v; v8bf h[2]; } u;
    u.h[0] = *(const v8bf*)(p);
    u.h[1] = *(const v8bf*)(p + 16);
    return u.v;
}

// ---------------------------------------------------------------------------
// bf16 WMMA GEMM: C[M,N] f32 = A[M,K]bf16 @ W[N,K]bf16^T
// Block tile 256(M) x 128(N): 8 waves, wave w owns rows [w*32, w*32+32)
// (2 M-tiles), all 8 N-tiles -> 16 accumulators (128 VGPRs).
// Weight panel (128 x K bf16) staged in LDS via the Tensor Data Mover.
// K loop is ping-pong unrolled by 2 (kIters is even): slice k computes with
// af0 while af1 loads, and vice versa -- no A-fragment register copies.
// grid.x = M/256, grid.y = N/128, dynamic LDS = 128*K*2 bytes.
// ---------------------------------------------------------------------------
__global__ __launch_bounds__(256) void lru_gemm_bf16(
    const __bf16* __restrict__ A, int lda,
    const __bf16* __restrict__ W, int ldb,
    float* __restrict__ C, int ldc, int kIters)
{
    extern __shared__ __bf16 wlds[];             // [128][K] bf16
    const int tid  = threadIdx.x;
    const int lane = tid & 31;
    const int wave = tid >> 5;                   // 0..7
    const int K    = kIters * 32;
    const int nB   = blockIdx.y * 128;

    // ---- Stage weight panel W[nB : nB+128][0:K] into LDS --------------------
#if __has_builtin(__builtin_amdgcn_tensor_load_to_lds)
    if (wave == 0) {
        // Tensor DMA descriptor (D#), ISA 8.3/8.4: 2-D tile, bf16 elements.
        unsigned long long ga = (unsigned long long)(size_t)(W + (size_t)nB * ldb);
        unsigned ldsa = (unsigned)(size_t)(&wlds[0]);
        unsigned uK = (unsigned)K;
        unsigned long long st = (unsigned long long)(unsigned)ldb;

        u32x4 g0;
        g0[0] = 1u;                                       // count=1, user mode
        g0[1] = ldsa;                                     // lds_addr
        g0[2] = (unsigned)ga;                             // global_addr[31:0]
        g0[3] = (unsigned)((ga >> 32) & 0x01FFFFFFu) | (2u << 30); // [56:32]|type=2

        unsigned w0 = (1u << 16);                         // data_size=1 (2 bytes)
        unsigned w1 = (uK & 0xFFFFu) << 16;               // tensor_dim0[15:0]
        unsigned w2 = (uK >> 16) | (128u << 16);          // dim0 hi | tensor_dim1=128
        unsigned w3 = (uK & 0xFFFFu) << 16;               // dim1 hi=0 | tile_dim0=K
        unsigned w4 = 128u;                               // tile_dim1=128, tile_dim2=0
        unsigned w5 = (unsigned)(st & 0xFFFFFFFFu);       // tensor_dim0_stride lo
        unsigned w6 = (unsigned)((st >> 32) & 0xFFFFu);   // stride hi | dim1_stride=0
        unsigned w7 = 0u;
        i32x8 g1 = { (int)w0, (int)w1, (int)w2, (int)w3,
                     (int)w4, (int)w5, (int)w6, (int)w7 };
        i32x4 g2 = { 0, 0, 0, 0 };                        // <=2D tensor: unused
        i32x4 g3 = { 0, 0, 0, 0 };
#if defined(__clang_major__) && (__clang_major__ >= 23)
        i32x8 g4 = { 0, 0, 0, 0, 0, 0, 0, 0 };
        __builtin_amdgcn_tensor_load_to_lds(g0, g1, g2, g3, g4, 0);
#else
        __builtin_amdgcn_tensor_load_to_lds(g0, g1, g2, g3, 0);
#endif
        __builtin_amdgcn_s_wait_tensorcnt(0);
    }
#else
    {   // Fallback: cooperative copy, 16B chunks
        const int vecPerRow = K / 8;
        for (int i = tid; i < 128 * vecPerRow; i += 256) {
            int r = i / vecPerRow, c = i % vecPerRow;
            *(v8bf*)&wlds[r * K + c * 8] =
                *(const v8bf*)&W[(size_t)(nB + r) * ldb + c * 8];
        }
    }
#endif
    __syncthreads();

    // ---- Main K loop: A from global (ping-pong buffered), B from LDS -------
    const int m0 = blockIdx.x * (128 * MT) + wave * (16 * MT);

    v8f acc[MT][8];
#pragma unroll
    for (int m = 0; m < MT; ++m)
#pragma unroll
        for (int i = 0; i < 8; ++i)
            acc[m][i] = (v8f){0.f,0.f,0.f,0.f,0.f,0.f,0.f,0.f};

    // One k-slice: 8 LDS B-frags into distinct regs (one clause), optional
    // A-frag load for the next slice + prefetch, then 16 back-to-back WMMAs.
    auto kstep = [&](v16bf (&afc)[MT], v16bf (&afn)[MT], int k0, bool loadNext) {
        v16bf bfr[8];
#pragma unroll
        for (int nt = 0; nt < 8; ++nt)
            bfr[nt] = lds_frag_b(wlds, K, nt * 16, k0, lane);
        if (loadNext) {
#pragma unroll
            for (int m = 0; m < MT; ++m)
                afn[m] = load_frag_a(A, lda, m0 + m * 16, k0 + 32, lane);
            // Pull the k+2 A panel toward the caches (global_prefetch_b8)
            const __bf16* pf = A + (size_t)(m0 + (lane & 15)) * lda + k0 + 64;
            __builtin_prefetch((const void*)pf, 0, 1);
        }
#pragma unroll
        for (int nt = 0; nt < 8; ++nt)
#pragma unroll
            for (int m = 0; m < MT; ++m)
                acc[m][nt] = __builtin_amdgcn_wmma_f32_16x16x32_bf16(
                    false, afc[m], false, bfr[nt], (short)0, acc[m][nt], false, false);
    };

    v16bf af0[MT], af1[MT];
#pragma unroll
    for (int m = 0; m < MT; ++m)
        af0[m] = load_frag_a(A, lda, m0 + m * 16, 0, lane);

    for (int kk = 0; kk < kIters; kk += 2) {
        kstep(af0, af1, kk * 32, kk + 1 < kIters);
        if (kk + 1 < kIters)
            kstep(af1, af0, (kk + 1) * 32, kk + 2 < kIters);
    }

    // C/D layout: VGPR r -> (M = mtile + r + 8*half, N = nB + nt*16 + (lane&15))
    const int half = lane >> 4;
    const int col  = lane & 15;
#pragma unroll
    for (int m = 0; m < MT; ++m)
#pragma unroll
        for (int nt = 0; nt < 8; ++nt)
#pragma unroll
            for (int r = 0; r < 8; ++r) {
                int row = m0 + m * 16 + r + half * 8;
                C[(size_t)row * ldc + nB + nt * 16 + col] = acc[m][nt][r];
            }
}

// ---------------------------------------------------------------------------
// Weight prep: Wb[512][128] = gamma-scaled [B_re; B_im] (bf16),
//              Wc[128][640] = [C_re | -C_im | D] (bf16)
// ---------------------------------------------------------------------------
__global__ __launch_bounds__(256) void lru_prep_weights(
    const float* __restrict__ gamma_log,
    const float* __restrict__ B_re, const float* __restrict__ B_im,
    const float* __restrict__ C_re, const float* __restrict__ C_im,
    const float* __restrict__ D,
    __bf16* __restrict__ Wb, __bf16* __restrict__ Wc)
{
    int i = blockIdx.x * blockDim.x + threadIdx.x;
    if (i < 512 * 128) {
        int n = i >> 7, h = i & 127;
        float g = __expf(gamma_log[n & 255]);
        float w = (n < 256) ? B_re[(size_t)n * 128 + h]
                            : B_im[(size_t)(n - 256) * 128 + h];
        Wb[i] = (__bf16)(g * w);
    } else {
        int j = i - 512 * 128;
        if (j < 128 * AK) {
            int o = j / AK, k = j % AK;
            float w;
            if (k < 256)        w =  C_re[(size_t)o * 256 + k];
            else if (k < 512)   w = -C_im[(size_t)o * 256 + (k - 256)];
            else                w =  D[(size_t)o * 128 + (k - 512)];
            Wc[j] = (__bf16)w;
        }
    }
}

// u f32 -> bf16 into Abuf columns [512..640)
__global__ __launch_bounds__(256) void lru_convert_u(
    const float* __restrict__ u, __bf16* __restrict__ Abuf)
{
    size_t i = (size_t)blockIdx.x * blockDim.x + threadIdx.x;
    if (i >= (size_t)BL * H_IN) return;
    size_t row = i >> 7;
    int h = (int)(i & 127);
    Abuf[row * AK + 512 + h] = (__bf16)u[i];
}

__device__ __forceinline__ void lru_lambda(const float* nu_log,
                                           const float* theta_log,
                                           int n, float& lr, float& li)
{
    float lm = __expf(-__expf(nu_log[n]));
    float th = __expf(theta_log[n]);
    lr = lm * __cosf(th);
    li = lm * __sinf(th);
}

// Pass 1: local scan per (batch, chunk). Overwrites Bu in place with the
// *local* shifted states (pre_t = x_{t-1}, x0=0), stores chunk-end state.
__global__ __launch_bounds__(256) void lru_scan_pass1(
    float* __restrict__ Bu, float* __restrict__ S,
    const float* __restrict__ nu_log, const float* __restrict__ theta_log)
{
    int n = threadIdx.x;           // 0..255
    int b = blockIdx.x;            // batch
    int c = blockIdx.y;            // chunk
    float lr, li; lru_lambda(nu_log, theta_log, n, lr, li);
    float xr = 0.f, xi = 0.f;
    size_t rowBase = ((size_t)b * SEQ + (size_t)c * CHUNK);
    for (int t = 0; t < CHUNK; ++t) {
        float* p = Bu + (rowBase + t) * 512;
        float br = p[n], bi = p[256 + n];
        p[n] = xr; p[256 + n] = xi;                 // local pre-state
        float nxr = lr * xr - li * xi + br;
        float nxi = lr * xi + li * xr + bi;
        xr = nxr; xi = nxi;
    }
    size_t so = (((size_t)b * NCHUNK + c) * 2) * 256;
    S[so + n] = xr; S[so + 256 + n] = xi;
}

// Pass 2: exclusive prefix across chunks (per batch): P[c] = l^64 P[c-1] + S[c-1]
__global__ __launch_bounds__(256) void lru_scan_pass2(
    const float* __restrict__ S, float* __restrict__ P,
    const float* __restrict__ nu_log, const float* __restrict__ theta_log)
{
    int n = threadIdx.x;
    int b = blockIdx.x;
    float lr, li; lru_lambda(nu_log, theta_log, n, lr, li);
    float cr = lr, ci = li;                      // lambda^(2^k) by squaring
#pragma unroll
    for (int k = 0; k < 6; ++k) {                // -> lambda^64
        float nr = cr * cr - ci * ci;
        ci = 2.f * cr * ci; cr = nr;
    }
    float pr = 0.f, pi = 0.f;
    for (int c = 0; c < NCHUNK; ++c) {
        size_t po = (((size_t)b * NCHUNK + c) * 2) * 256;
        P[po + n] = pr; P[po + 256 + n] = pi;
        float sr = S[po + n], si = S[po + 256 + n];
        float npr = cr * pr - ci * pi + sr;
        float npi = cr * pi + ci * pr + si;
        pr = npr; pi = npi;
    }
}

// Pass 3: global pre = lambda^tl * P + pre_local; emit bf16 into Abuf[0..512)
__global__ __launch_bounds__(256) void lru_scan_pass3(
    const float* __restrict__ Bu, const float* __restrict__ P,
    __bf16* __restrict__ Abuf,
    const float* __restrict__ nu_log, const float* __restrict__ theta_log)
{
    int n = threadIdx.x;
    int b = blockIdx.x;
    int c = blockIdx.y;
    float lr, li; lru_lambda(nu_log, theta_log, n, lr, li);
    size_t po = (((size_t)b * NCHUNK + c) * 2) * 256;
    float pr = P[po + n], pi = P[po + 256 + n];
    float wr = 1.f, wi = 0.f;                    // lambda^tl
    size_t rowBase = ((size_t)b * SEQ + (size_t)c * CHUNK);
    for (int t = 0; t < CHUNK; ++t) {
        const float* p = Bu + (rowBase + t) * 512;
        float gr = wr * pr - wi * pi + p[n];
        float gi = wr * pi + wi * pr + p[256 + n];
        __bf16* a = Abuf + (rowBase + t) * AK;
        a[n] = (__bf16)gr;
        a[256 + n] = (__bf16)gi;
        float nwr = wr * lr - wi * li;
        float nwi = wr * li + wi * lr;
        wr = nwr; wi = nwi;
    }
}

// ---------------------------------------------------------------------------
extern "C" void kernel_launch(void* const* d_in, const int* in_sizes, int n_in,
                              void* d_out, int out_size, void* d_ws, size_t ws_size,
                              hipStream_t stream) {
    const float* u         = (const float*)d_in[0];
    const float* nu_log    = (const float*)d_in[1];
    const float* theta_log = (const float*)d_in[2];
    const float* gamma_log = (const float*)d_in[3];
    const float* B_re      = (const float*)d_in[4];
    const float* B_im      = (const float*)d_in[5];
    const float* C_re      = (const float*)d_in[6];
    const float* C_im      = (const float*)d_in[7];
    const float* Dm        = (const float*)d_in[8];
    float* out             = (float*)d_out;

    // Workspace carve-up
    char* ws = (char*)d_ws;
    __bf16* Abuf = (__bf16*)ws;                 // BL x 640 bf16  (83.9 MB)
    ws += (size_t)BL * AK * sizeof(__bf16);
    float* Bu = (float*)ws;                     // BL x 512 f32   (134.2 MB)
    ws += (size_t)BL * 512 * sizeof(float);
    __bf16* Wb = (__bf16*)ws;                   // 512 x 128 bf16
    ws += (size_t)512 * 128 * sizeof(__bf16);
    __bf16* Wc = (__bf16*)ws;                   // 128 x 640 bf16
    ws += (size_t)128 * AK * sizeof(__bf16);
    float* S = (float*)ws;                      // B x NCHUNK x 2 x 256 f32
    ws += (size_t)BATCH * NCHUNK * 2 * 256 * sizeof(float);
    float* P = (float*)ws;
    ws += (size_t)BATCH * NCHUNK * 2 * 256 * sizeof(float);

    // 1) Weight prep + u -> bf16 (into fused activation layout)
    {
        int total = 512 * 128 + 128 * AK;
        lru_prep_weights<<<(total + 255) / 256, 256, 0, stream>>>(
            gamma_log, B_re, B_im, C_re, C_im, Dm, Wb, Wc);
        size_t n = (size_t)BL * H_IN;
        lru_convert_u<<<(unsigned)((n + 255) / 256), 256, 0, stream>>>(u, Abuf);
    }

    // 2) Projection GEMM: Bu[BL,512] = u_bf16[BL,128] @ Wb[512,128]^T  (K=128)
    lru_gemm_bf16<<<dim3(BL / (128 * MT), 512 / 128), 256,
                    128 * 128 * sizeof(__bf16), stream>>>(
        Abuf + 512, AK, Wb, 128, Bu, 512, 128 / 32);

    // 3) Chunked complex scan (3 passes)
    lru_scan_pass1<<<dim3(BATCH, NCHUNK), 256, 0, stream>>>(Bu, S, nu_log, theta_log);
    lru_scan_pass2<<<BATCH, 256, 0, stream>>>(S, P, nu_log, theta_log);
    lru_scan_pass3<<<dim3(BATCH, NCHUNK), 256, 0, stream>>>(Bu, P, Abuf, nu_log, theta_log);

    // 4) Fused output GEMM: y[BL,128] = [pre_re|pre_im|u] @ [C_re|-C_im|D]^T (K=640)
    lru_gemm_bf16<<<dim3(BL / (128 * MT), H_OUT / 128), 256,
                    128 * AK * sizeof(__bf16), stream>>>(
        Abuf, AK, Wc, AK, out, H_OUT, AK / 32);
}
